// CharRNN_86792699118075
// MI455X (gfx1250) — matrix operand order repaired and checked by reading, not compiled, not run on hardware
//
#include <hip/hip_runtime.h>
#include <hip/hip_bf16.h>

// CharRNN on MI455X (gfx1250), wave32 + WMMA bf16 path + TDM staging.
// B=64, S=512, E=512, H=1024, V=256
#define B_ 64
#define S_ 512
#define E_ 512
#define H_ 1024
#define V_ 256

typedef __attribute__((ext_vector_type(16))) __bf16        v16bf;
typedef __attribute__((ext_vector_type(8)))  float         v8f;
typedef __attribute__((ext_vector_type(4)))  unsigned int  u32x4;
typedef __attribute__((ext_vector_type(8)))  int           i32x8;
typedef __attribute__((ext_vector_type(4)))  int           i32x4;

union Frag32B { v16bf v; u32x4 u[2]; };   // 32-byte WMMA 16-bit operand fragment

// Pick the best available CDNA5 LDS-staging path (device pass only).
#if defined(__HIP_DEVICE_COMPILE__)
#  if __has_builtin(__builtin_amdgcn_tensor_load_to_lds) && \
      __has_builtin(__builtin_amdgcn_s_wait_tensorcnt)
#    define USE_TDM 1
#  elif __has_builtin(__builtin_amdgcn_global_load_async_to_lds_b128) && \
        __has_builtin(__builtin_amdgcn_s_wait_asynccnt)
#    define USE_ASYNC_COPY 1
#  endif
#endif

__device__ __forceinline__ v8f wmma_bf16(v16bf a, v16bf b, v8f c) {
    // D = A(16x32 bf16) x B(32x16 bf16) + C(16x16 f32)
    return __builtin_amdgcn_wmma_f32_16x16x32_bf16(
        /*neg_a=*/false, a, /*neg_b=*/false, b,
        /*c_mod=*/(short)0, c, /*reuse_a=*/false, /*reuse_b=*/false);
}

// ---------------------------------------------------------------------------
// Prep: zero grid-sync counter, bf16-convert initial hidden, and pack W_h /
// W_o into WMMA B-fragment-ready layout:
//   packed[((ntile*32 + kt)*32 + lane)*16 + i] = W[K][N],
//   K = kt*32 + (i>>3)*16 + (lane>>4)*8 + (i&7),  N = ntile*16 + (lane&15)
// ---------------------------------------------------------------------------
__global__ void prep_kernel(const float* __restrict__ hidden,
                            const float* __restrict__ W_h,
                            const float* __restrict__ W_o,
                            __bf16* __restrict__ whp,
                            __bf16* __restrict__ wop,
                            __bf16* __restrict__ Hping,
                            unsigned int* __restrict__ sync) {
    const unsigned idx = blockIdx.x * blockDim.x + threadIdx.x;   // [0, 1<<20)
    if (idx == 0) *sync = 0u;
    {
        const unsigned i = idx & 15u, lane = (idx >> 4) & 31u;
        const unsigned kt = (idx >> 9) & 31u, nt = idx >> 14;      // nt < 64
        const unsigned K = kt * 32u + (i >> 3) * 16u + (lane >> 4) * 8u + (i & 7u);
        const unsigned N = nt * 16u + (lane & 15u);
        whp[idx] = (__bf16)W_h[(size_t)K * H_ + N];
    }
    if (idx < (1u << 18)) {                                        // 1024*256
        const unsigned i = idx & 15u, lane = (idx >> 4) & 31u;
        const unsigned kt = (idx >> 9) & 31u, nt = idx >> 14;      // nt < 16
        const unsigned K = kt * 32u + (i >> 3) * 16u + (lane >> 4) * 8u + (i & 7u);
        const unsigned N = nt * 16u + (lane & 15u);
        wop[idx] = (__bf16)W_o[(size_t)K * V_ + N];
    }
    if (idx < (unsigned)(B_ * H_)) Hping[idx] = (__bf16)hidden[idx];
}

// ---------------------------------------------------------------------------
// proj_table[256][1024] = embedding[256][512] @ W_e[512][1024]
// One wave per 16x16 tile, K=512 (16 WMMAs). One-time cost; element loads OK.
// ---------------------------------------------------------------------------
__global__ void proj_table_kernel(const float* __restrict__ emb,
                                  const float* __restrict__ W_e,
                                  float* __restrict__ proj) {
    const unsigned tid = threadIdx.x, lane = tid & 31u, wave = tid >> 5;
    const unsigned tile = blockIdx.x * 4u + wave;   // 0..1023
    const unsigned mt = tile >> 6, nt = tile & 63u; // mt<16 (M=256), nt<64 (N=1024)
    const unsigned hi = lane >> 4, lo = lane & 15u;
    const unsigned arow = mt * 16u + lo;
    const unsigned bcol = nt * 16u + lo;
    v8f acc = {};
    for (unsigned kt = 0; kt < 16; ++kt) {
        const unsigned kb = kt * 32u;
        v16bf a, b;
#pragma unroll
        for (unsigned i = 0; i < 16; ++i) {
            const unsigned ka = kb + (i >> 3) * 16u + hi * 8u + (i & 7u);
            a[i] = (__bf16)emb[(size_t)arow * E_ + ka];
            const unsigned kB = kb + (i >> 3) * 16u + hi * 8u + (i & 7u);
            b[i] = (__bf16)W_e[(size_t)kB * H_ + bcol];
        }
        acc = wmma_bf16(a, b, acc);
    }
#pragma unroll
    for (unsigned r = 0; r < 8; ++r) {
        const unsigned row = mt * 16u + hi * 8u + r;
        proj[(size_t)row * H_ + bcol] = acc[r];
    }
}

// ---------------------------------------------------------------------------
// Persistent recurrence: 64 WGs x 128 threads (4 waves). WG g owns hidden
// columns [g*16, g*16+16). W_h slice lives in LDS in fragment layout (staged
// by TDM when available). Each step:
//   H_new = tanh(H_prev @ W_h + proj_table[x[:,t]]),  then grid barrier.
// ---------------------------------------------------------------------------
__global__ void __launch_bounds__(128, 1)
rnn_step_kernel(const int* __restrict__ x,
                const float* __restrict__ proj,
                const __bf16* __restrict__ whp,
                __bf16* __restrict__ Hping, __bf16* __restrict__ Hpong,
                __bf16* __restrict__ outs,            // [S][B][H] bf16
                float* __restrict__ finalH,           // [B][H] f32 (d_out tail)
                unsigned int* __restrict__ sync) {
    __shared__ __align__(16) __bf16 ldsB[32u * 32u * 16u];   // 32 KB slice of W_h
    __shared__ int ldsx[B_];
    const unsigned tid = threadIdx.x, lane = tid & 31u, wave = tid >> 5;
    const unsigned wg = blockIdx.x;                   // 0..63

    // ---- Stage this WG's packed W_h slice (16384 bf16 = 32 KB) into LDS ----
#if defined(USE_TDM)
    if (tid < 32u) {   // wave 0 issues one TDM op; TENSORcnt is per-wave
        const unsigned long long ga =
            (unsigned long long)(uintptr_t)(whp + (size_t)wg * 16384u);
        const unsigned ldsOff = (unsigned)(uintptr_t)(void*)ldsB;
        u32x4 g0;
        g0.x = 1u;                                     // count=1 user descriptor
        g0.y = ldsOff;                                 // lds_addr [63:32]
        g0.z = (unsigned)(ga & 0xffffffffu);           // global_addr [95:64]
        g0.w = (unsigned)((ga >> 32) & 0x01ffffffu)    // global_addr [120:96]
             | (2u << 30);                             // type=2 ("image")
        i32x8 g1;
        g1[0] = 0x10000;                 // data_size=1 (2-byte elements)
        g1[1] = (int)(16384u << 16);     // tensor_dim0[15:0] @ bits [63:48]
        g1[2] = 0;                       // tensor_dim0[31:16]=0, tensor_dim1=0
        g1[3] = (int)(16384u << 16);     // tile_dim0 = 16384 @ bits [127:112]
        g1[4] = 0;                       // tile_dim1=0 (1-D), tile_dim2=0
        g1[5] = 16384;                   // tensor_dim0_stride[31:0]
        g1[6] = 0;
        g1[7] = 0;
        i32x4 z4 = {0, 0, 0, 0};
#if __clang_major__ >= 23
        i32x8 z8 = {};
        __builtin_amdgcn_tensor_load_to_lds(g0, g1, z4, z4, z8, 0);
#else
        __builtin_amdgcn_tensor_load_to_lds(g0, g1, z4, z4, 0);
#endif
        __builtin_amdgcn_s_wait_tensorcnt(0);
    }
#elif defined(USE_ASYNC_COPY)
    {
        const u32x4* src = (const u32x4*)(whp + (size_t)wg * 16384u);
        u32x4* dst = (u32x4*)ldsB;
#pragma unroll
        for (unsigned j = 0; j < 16; ++j) {
            __builtin_amdgcn_global_load_async_to_lds_b128(
                (__attribute__((address_space(1))) const void*)(src + tid + j * 128u),
                (__attribute__((address_space(3))) void*)(dst + tid + j * 128u),
                0, 0);
        }
        __builtin_amdgcn_s_wait_asynccnt(0);
    }
#else
    {
        const u32x4* src = (const u32x4*)(whp + (size_t)wg * 16384u);
        u32x4* dst = (u32x4*)ldsB;
#pragma unroll
        for (unsigned j = 0; j < 16; ++j)
            dst[tid + j * 128u] = src[tid + j * 128u];
    }
#endif
    __syncthreads();

    const unsigned hi = lane >> 4, lo = lane & 15u;
    const unsigned m0 = wave * 16u;                   // batch-row tile base
    const unsigned arow = m0 + lo;                    // A-fragment row (batch)
    const unsigned col = wg * 16u + lo;               // owned hidden column

    for (unsigned t = 0; t < S_; ++t) {
        __syncthreads();
        if (tid < (unsigned)B_) ldsx[tid] = x[(size_t)tid * S_ + t];
        __syncthreads();

        const __bf16* cur = (t & 1u) ? Hpong : Hping;
        __bf16* nxt = (t & 1u) ? Hping : Hpong;

        // Two accumulators: halves the WMMA->WMMA dependent chain.
        v8f acc0 = {}, acc1 = {};
        const u32x4* arowp = (const u32x4*)(cur + (size_t)arow * H_);
#pragma unroll 2
        for (unsigned kt = 0; kt < 32; kt += 2) {
            {
                const unsigned kb = kt * 32u;
                Frag32B a, b;
                a.u[0] = arowp[(kb + hi * 8u) >> 3];
                a.u[1] = arowp[(kb + 16u + hi * 8u) >> 3];
                const u32x4* bp = (const u32x4*)(ldsB + ((size_t)kt * 32u + lane) * 16u);
                b.u[0] = bp[0];
                b.u[1] = bp[1];
                acc0 = wmma_bf16(a.v, b.v, acc0);
            }
            {
                const unsigned kb = (kt + 1u) * 32u;
                Frag32B a, b;
                a.u[0] = arowp[(kb + hi * 8u) >> 3];
                a.u[1] = arowp[(kb + 16u + hi * 8u) >> 3];
                const u32x4* bp = (const u32x4*)(ldsB + ((size_t)(kt + 1u) * 32u + lane) * 16u);
                b.u[0] = bp[0];
                b.u[1] = bp[1];
                acc1 = wmma_bf16(a.v, b.v, acc1);
            }
        }
        const v8f acc = acc0 + acc1;

#pragma unroll
        for (unsigned r = 0; r < 8; ++r) {
            const unsigned brow = m0 + hi * 8u + r;       // batch index
            const float xp = proj[(size_t)ldsx[brow] * H_ + col];
            const float h = tanhf(acc[r] + xp);
            const __bf16 hb = (__bf16)h;
            outs[(size_t)t * (B_ * H_) + (size_t)brow * H_ + col] = hb;
            nxt[(size_t)brow * H_ + col] = hb;
            if (t == S_ - 1) finalH[(size_t)brow * H_ + col] = h;
        }

        // ---- device-scope grid barrier (all 64 WGs resident) ----
        __threadfence();        // release this thread's stores to agent scope
        __syncthreads();
        if (tid == 0) {
            atomicAdd(sync, 1u);
            const unsigned target = (t + 1u) * 64u;
            while (__hip_atomic_load(sync, __ATOMIC_ACQUIRE,
                                     __HIP_MEMORY_SCOPE_AGENT) < target)
                __builtin_amdgcn_s_sleep(2);
        }
        __syncthreads();
        __threadfence();        // acquire: invalidate stale near caches
    }
}

// ---------------------------------------------------------------------------
// logits[b][t][v] = outs[t][b][:] @ W_o   (A rows r = t*64+b are contiguous)
// One wave per 16x16 tile; K=1024 (32 WMMAs, 2 accumulators).
// ---------------------------------------------------------------------------
__global__ void logits_kernel(const __bf16* __restrict__ outs,
                              const __bf16* __restrict__ wop,
                              float* __restrict__ logits) {
    const unsigned tid = threadIdx.x, lane = tid & 31u, wave = tid >> 5;
    const unsigned tile = blockIdx.x * 4u + wave;     // 0..32767
    const unsigned nt = tile & 15u, mt = tile >> 4;   // nt<16, mt<2048
    const unsigned hi = lane >> 4, lo = lane & 15u;
    const unsigned arow = mt * 16u + lo;              // row = t*64 + b
    v8f acc0 = {}, acc1 = {};
    const u32x4* arowp = (const u32x4*)(outs + (size_t)arow * H_);
    const u32x4* bbase = (const u32x4*)(wop + (size_t)nt * 16384u);
#pragma unroll 2
    for (unsigned kt = 0; kt < 32; kt += 2) {
        {
            const unsigned kb = kt * 32u;
            Frag32B a, b;
            a.u[0] = arowp[(kb + hi * 8u) >> 3];
            a.u[1] = arowp[(kb + 16u + hi * 8u) >> 3];
            const u32x4* bp = bbase + ((size_t)kt * 32u + lane) * 2u;
            b.u[0] = bp[0];
            b.u[1] = bp[1];
            acc0 = wmma_bf16(a.v, b.v, acc0);
        }
        {
            const unsigned kb = (kt + 1u) * 32u;
            Frag32B a, b;
            a.u[0] = arowp[(kb + hi * 8u) >> 3];
            a.u[1] = arowp[(kb + 16u + hi * 8u) >> 3];
            const u32x4* bp = bbase + ((size_t)(kt + 1u) * 32u + lane) * 2u;
            b.u[0] = bp[0];
            b.u[1] = bp[1];
            acc1 = wmma_bf16(a.v, b.v, acc1);
        }
    }
    const v8f acc = acc0 + acc1;
    const unsigned colv = nt * 16u + lo;
#pragma unroll
    for (unsigned r = 0; r < 8; ++r) {
        const unsigned R = mt * 16u + hi * 8u + r;    // = t*64 + b
        const unsigned tt = R >> 6, bb = R & 63u;
        logits[(size_t)bb * (S_ * V_) + (size_t)tt * V_ + colv] = acc[r];
    }
}

// ---------------------------------------------------------------------------
extern "C" void kernel_launch(void* const* d_in, const int* in_sizes, int n_in,
                              void* d_out, int out_size, void* d_ws, size_t ws_size,
                              hipStream_t stream) {
    const int*   x   = (const int*)d_in[0];     // [64][512]
    const float* hid = (const float*)d_in[1];   // [64][1024]
    const float* emb = (const float*)d_in[2];   // [256][512]
    const float* W_e = (const float*)d_in[3];   // [512][1024]
    const float* W_h = (const float*)d_in[4];   // [1024][1024]
    const float* W_o = (const float*)d_in[5];   // [1024][256]
    float* out = (float*)d_out;                 // logits [64][512][256] ++ finalH [64][1024]

    char* ws = (char*)d_ws;
    unsigned int* sync = (unsigned int*)ws;                       // 256 B reserved
    float*  proj  = (float*)(ws + 256);                           // 1 MB
    __bf16* Hping = (__bf16*)(ws + 256 + 1048576);                // 128 KB
    __bf16* Hpong = Hping + (size_t)B_ * H_;                      // 128 KB
    __bf16* whp   = (__bf16*)(ws + 1310976);                      // 2 MB packed W_h
    __bf16* wop   = (__bf16*)(ws + 3408128);                      // 512 KB packed W_o
    __bf16* outs  = (__bf16*)(ws + 3932416);                      // 64 MB [S][B][H]

    float* logits = out;
    float* finalH = out + (size_t)B_ * S_ * V_;

    prep_kernel      <<<4096, 256, 0, stream>>>(hid, W_h, W_o, whp, wop, Hping, sync);
    proj_table_kernel<<<256, 128, 0, stream>>>(emb, W_e, proj);
    rnn_step_kernel  <<<64, 128, 0, stream>>>(x, proj, whp, Hping, Hpong, outs, finalH, sync);
    logits_kernel    <<<8192, 128, 0, stream>>>(outs, wop, logits);
}
